// MultiheadAttentionWithAttention_PyTorch_69896297775143
// MI455X (gfx1250) — compile-verified, matrix operand order checked by reading, not run on hardware
//
#include <hip/hip_runtime.h>

typedef __bf16 bf16;
typedef __attribute__((ext_vector_type(16))) __bf16 v16bf;
typedef __attribute__((ext_vector_type(8)))  __bf16 v8bf;
typedef __attribute__((ext_vector_type(8)))  float  v8f;
typedef __attribute__((ext_vector_type(4)))  float  v4f;
typedef __attribute__((ext_vector_type(4)))  unsigned int v4u;
typedef __attribute__((ext_vector_type(8)))  int    v8i;
typedef __attribute__((ext_vector_type(4)))  int    v4i;

#if __has_builtin(__builtin_amdgcn_tensor_load_to_lds)
#define HAVE_TDM 1
#else
#define HAVE_TDM 0
#endif

// ---------------------------------------------------------------------------
// WMMA helpers (CDNA5 16x16x32 bf16, f32 accumulate)
// A operand (16x32, 16-bit): lane l holds row M = l&15; with g = l>>4 the lane
// holds K chunks [8g, 8g+8) in v0..3 and [16+8g, 16+8g+8) in v4..7.
// B operand loaded as B^T rows with the same per-lane pattern (lane = N col).
// C/D: lane n = l&15 is column; VGPR r holds row M = 8*(l>>4) + r.
// ---------------------------------------------------------------------------
union FragAB { v16bf v; v8bf h[2]; };

__device__ inline v16bf load_ab(const bf16* __restrict__ rowptr, int g) {
  FragAB f;
  f.h[0] = *(const v8bf*)(rowptr + 8 * g);
  f.h[1] = *(const v8bf*)(rowptr + 16 + 8 * g);
  return f.v;
}

__device__ inline v8f wmma_bf16(v16bf a, v16bf b, v8f c) {
  return __builtin_amdgcn_wmma_f32_16x16x32_bf16(
      /*neg_a=*/false, a, /*neg_b=*/false, b,
      /*c_mod=*/(short)0, c, /*reuse_a=*/false, /*reuse_b=*/false);
}

__device__ inline v8f zero8() {
  v8f z = {0.f, 0.f, 0.f, 0.f, 0.f, 0.f, 0.f, 0.f};
  return z;
}

// ---------------------------------------------------------------------------
// Tensor Data Mover: DMA a 64-row x 128-col bf16 tile of W into LDS.
// D# per CDNA5 ISA ch.8: group0 = {count, lds_addr, global_addr, type=2},
// group1 = {data_size=2B, tensor_dim0=K, tensor_dim1=N, tile_dim0=128,
//           tile_dim1=64, tensor_dim0_stride=K}; groups 2/3 zero (2-D).
// ---------------------------------------------------------------------------
#define KCH 128  // K-chunk staged per TDM transfer

#if HAVE_TDM
__device__ inline void tdm_load_w_tile(const bf16* W, int row0, int kbase,
                                       unsigned lds_byte_off, int K, int N) {
  unsigned long long ga =
      (unsigned long long)(const void*)(W + (size_t)row0 * K + kbase);
  v4u g0;
  g0.x = 1u;                                   // count=1, user descriptor
  g0.y = lds_byte_off;                         // LDS destination (bytes)
  g0.z = (unsigned)(ga & 0xFFFFFFFFu);         // global_addr[31:0]
  g0.w = (unsigned)((ga >> 32) & 0x01FFFFFFu)  // global_addr[56:32]
         | (2u << 30);                         // type=2 ("image")
  v8i g1;
  g1[0] = 0x00010000;                              // data_size=1 (2 bytes)
  g1[1] = (int)(((unsigned)K & 0xFFFFu) << 16);    // tensor_dim0[15:0]
  g1[2] = (int)(((unsigned)K >> 16) |
                (((unsigned)N & 0xFFFFu) << 16));  // dim0 hi | dim1 lo
  g1[3] = (int)(((unsigned)N >> 16) |
                ((unsigned)KCH << 16));            // dim1 hi | tile_dim0
  g1[4] = 64;                                      // tile_dim1 (rows)
  g1[5] = K;                                       // tensor_dim0_stride[31:0]
  g1[6] = 0;
  g1[7] = 0;
  v4i z4 = {0, 0, 0, 0};
#if defined(__clang_major__) && (__clang_major__ >= 23)
  v8i z8 = {0, 0, 0, 0, 0, 0, 0, 0};
  __builtin_amdgcn_tensor_load_to_lds(g0, g1, z4, z4, z8, 0);
#else
  __builtin_amdgcn_tensor_load_to_lds(g0, g1, z4, z4, 0);
#endif
}
#endif

// ---------------------------------------------------------------------------
// f32 -> bf16 conversion (8 elements / thread, packed 16B stores)
// ---------------------------------------------------------------------------
__global__ __launch_bounds__(256) void cvt_bf16(const float* __restrict__ src,
                                                bf16* __restrict__ dst, int n) {
  int i = (blockIdx.x * 256 + threadIdx.x) * 8;
  if (i + 8 <= n) {
    v4f a = *(const v4f*)(src + i);
    v4f b = *(const v4f*)(src + i + 4);
    v8bf o;
    o[0] = (bf16)a[0]; o[1] = (bf16)a[1]; o[2] = (bf16)a[2]; o[3] = (bf16)a[3];
    o[4] = (bf16)b[0]; o[5] = (bf16)b[1]; o[6] = (bf16)b[2]; o[7] = (bf16)b[3];
    *(v8bf*)(dst + i) = o;
  }
}

// ---------------------------------------------------------------------------
// out = X @ W^T + bias.  X:[M,K] bf16 row-major, W:[N,K] bf16 row-major.
// Block: 256 threads = 8 waves; wave owns rows r0..r0+15, cols c0..c0+63.
// W tile (64 x KCH) is shared by all 8 waves: staged in LDS by the TDM,
// double-buffered, TENSORcnt-tracked (DMA of chunk c+1 overlaps WMMA on c).
// MODE 0: f32 row-major store.  MODE 1: bf16 row-major store.
// MODE 2: bf16 transposed-per-head store: out[(b*1024 + c)*1024 + s]
//         where global row = b*1024 + s  (B=batch, T=1024, N=1024).
// ---------------------------------------------------------------------------
template <int MODE>
__global__ __launch_bounds__(256)
void gemm_xwT(const bf16* __restrict__ X, const bf16* __restrict__ W,
              const float* __restrict__ bias, void* __restrict__ out,
              int M, int N, int K) {
  // Only static LDS object in this kernel -> LDS offset 0; buf1 at 16384 B.
  __shared__ bf16 sW[2][64][KCH];  // 32 KB double buffer

  const int tid  = threadIdx.x;
  const int wave = tid >> 5;
  const int lane = tid & 31;
  const int ln   = lane & 15;
  const int g    = lane >> 4;
  const int r0   = blockIdx.y * 128 + wave * 16;
  const int c0   = blockIdx.x * 64;

  v8f acc[4];
#pragma unroll
  for (int t = 0; t < 4; ++t) acc[t] = zero8();

  const bf16* xrow = X + (size_t)(r0 + ln) * K;
  const int nch = K / KCH;

#if HAVE_TDM
  if (wave == 0) tdm_load_w_tile(W, c0, 0, 0u, K, N);
#endif

  for (int c = 0; c < nch; ++c) {
#if HAVE_TDM
    if (wave == 0) {
      if (c + 1 < nch) {
        tdm_load_w_tile(W, c0, (c + 1) * KCH,
                        (unsigned)(((c + 1) & 1) * sizeof(bf16) * 64 * KCH),
                        K, N);
        __builtin_amdgcn_s_wait_tensorcnt((short)1);  // chunk c landed
      } else {
        __builtin_amdgcn_s_wait_tensorcnt((short)0);
      }
    }
    __syncthreads();
#else
    // Fallback: cooperative vector copy of chunk c into LDS.
    for (int i = tid; i < 64 * KCH / 8; i += 256) {
      const int row = i / (KCH / 8), ch8 = i % (KCH / 8);
      ((v8bf*)&sW[c & 1][0][0])[i] =
          *(const v8bf*)(W + (size_t)(c0 + row) * K + c * KCH + ch8 * 8);
    }
    __syncthreads();
#endif

    const bf16* wt = &sW[c & 1][0][0];
#pragma unroll
    for (int kk = 0; kk < KCH; kk += 32) {
      __builtin_prefetch(xrow + c * KCH + kk + 128, 0, 0);
      v16bf a = load_ab(xrow + c * KCH + kk, g);
#pragma unroll
      for (int t = 0; t < 4; ++t) {
        v16bf b = load_ab(wt + (t * 16 + ln) * KCH + kk, g);
        acc[t] = wmma_bf16(a, b, acc[t]);
      }
    }
    __syncthreads();  // tile consumed before TDM overwrites this buffer
  }

#pragma unroll
  for (int t = 0; t < 4; ++t) {
    const int c  = c0 + t * 16 + ln;
    const float bv = bias[c];
    if (MODE == 0) {
      float* o = (float*)out;
#pragma unroll
      for (int r = 0; r < 8; ++r)
        o[(size_t)(r0 + 8 * g + r) * N + c] = acc[t][r] + bv;
    } else if (MODE == 1) {
      bf16* o = (bf16*)out;
#pragma unroll
      for (int r = 0; r < 8; ++r)
        o[(size_t)(r0 + 8 * g + r) * N + c] = (bf16)(acc[t][r] + bv);
    } else {
      bf16* o = (bf16*)out;
      const int bb = r0 >> 10;            // batch index
      const int sb = (r0 & 1023) + 8 * g; // 8 consecutive sequence positions
      v8bf pk;
#pragma unroll
      for (int r = 0; r < 8; ++r) pk[r] = (bf16)(acc[t][r] + bv);
      *(v8bf*)(o + ((size_t)(bb * 1024 + c)) * 1024 + sb) = pk;
    }
  }
}

// ---------------------------------------------------------------------------
// Attention: one workgroup per (batch, 16-query block).
// Qb:[B*T,1024] bf16, Kb:[B*T,1024] bf16 (head cols h*64..), Vt:[(b*1024+c),s].
// Ytmp:[B*T,1024] bf16 (pre-projection y), attOut:[B,T,T_MEM] f32 (head mean).
// ---------------------------------------------------------------------------
#define SM_SQ    0
#define SM_SP    (32 * 1024)
#define SM_ATT   (64 * 1024)
#define SM_YP    (128 * 1024)
#define SM_RMAX  (SM_YP + 8 * 1024)
#define SM_RSUM  (SM_RMAX + 512)
#define SM_RINV  (SM_RSUM + 512)
#define SM_TOTAL (SM_RINV + 64)

__global__ __launch_bounds__(256)
void attention_kernel(const bf16* __restrict__ Qb, const bf16* __restrict__ Kb,
                      const bf16* __restrict__ Vt, bf16* __restrict__ Ytmp,
                      float* __restrict__ attOut) {
  extern __shared__ char smem[];
  bf16*  sQ      = (bf16*)(smem + SM_SQ);    // [16][1024] bf16
  bf16*  sP      = (bf16*)(smem + SM_SP);    // [16][1024] bf16, exp(scores)
  float* sAtt    = (float*)(smem + SM_ATT);  // [16][1024] f32, head-sum of att
  float* sYp     = (float*)(smem + SM_YP);   // [2][16][64] f32, partial y
  float* sRedMax = (float*)(smem + SM_RMAX); // [8][16]
  float* sRedSum = (float*)(smem + SM_RSUM); // [8][16]
  float* sRowInv = (float*)(smem + SM_RINV); // [16]

  const int tid  = threadIdx.x;
  const int wave = tid >> 5;
  const int lane = tid & 31;
  const int ln   = lane & 15;
  const int g    = lane >> 4;
  const int bIdx = blockIdx.x >> 6;
  const int t0   = (blockIdx.x & 63) * 16;

  // Stage 16x1024 Q rows into LDS; zero the attention-sum accumulator.
  {
    const v8bf* src = (const v8bf*)(Qb + ((size_t)(bIdx * 1024 + t0)) * 1024);
    for (int i = tid; i < 2048; i += 256) ((v8bf*)sQ)[i] = src[i];
    v4f z = {0.f, 0.f, 0.f, 0.f};
    for (int i = tid; i < 4096; i += 256) ((v4f*)sAtt)[i] = z;
  }
  __syncthreads();

  for (int h = 0; h < 16; ++h) {
    // -------- scores: S = (Q_h K_h^T) / 8, wave owns 128 keys --------
    const bf16* qrow = sQ + ln * 1024 + h * 64;
    v16bf aq0 = load_ab(qrow, g);
    v16bf aq1 = load_ab(qrow + 32, g);

    v8f sc[8];
#pragma unroll
    for (int kt = 0; kt < 8; ++kt) {
      const int s0 = (wave * 8 + kt) * 16;
      const bf16* krow = Kb + ((size_t)(bIdx * 1024 + s0 + ln)) * 1024 + h * 64;
      v16bf b0 = load_ab(krow, g);
      v16bf b1 = load_ab(krow + 32, g);
      v8f c = zero8();
      c = wmma_bf16(aq0, b0, c);
      c = wmma_bf16(aq1, b1, c);
#pragma unroll
      for (int r = 0; r < 8; ++r) c[r] *= 0.125f; // 1/sqrt(64)
      sc[kt] = c;
    }

    // -------- row max (over keys) --------
    float lmax[8];
#pragma unroll
    for (int r = 0; r < 8; ++r) {
      float m = sc[0][r];
#pragma unroll
      for (int kt = 1; kt < 8; ++kt) m = fmaxf(m, sc[kt][r]);
#pragma unroll
      for (int off = 1; off < 16; off <<= 1) m = fmaxf(m, __shfl_xor(m, off, 32));
      lmax[r] = m;
    }
    if (ln == 0) {
#pragma unroll
      for (int r = 0; r < 8; ++r) sRedMax[wave * 16 + 8 * g + r] = lmax[r];
    }
    __syncthreads();
    float rmax[8];
#pragma unroll
    for (int r = 0; r < 8; ++r) {
      float m = sRedMax[8 * g + r];
#pragma unroll
      for (int w = 1; w < 8; ++w) m = fmaxf(m, sRedMax[w * 16 + 8 * g + r]);
      rmax[r] = m;
    }

    // -------- exp + row sum --------
#pragma unroll
    for (int kt = 0; kt < 8; ++kt)
#pragma unroll
      for (int r = 0; r < 8; ++r)
        sc[kt][r] = exp2f((sc[kt][r] - rmax[r]) * 1.4426950408889634f);

    float lsum[8];
#pragma unroll
    for (int r = 0; r < 8; ++r) {
      float s = 0.f;
#pragma unroll
      for (int kt = 0; kt < 8; ++kt) s += sc[kt][r];
#pragma unroll
      for (int off = 1; off < 16; off <<= 1) s += __shfl_xor(s, off, 32);
      lsum[r] = s;
    }
    if (ln == 0) {
#pragma unroll
      for (int r = 0; r < 8; ++r) sRedSum[wave * 16 + 8 * g + r] = lsum[r];
    }
    __syncthreads();
    float rinv[8];
#pragma unroll
    for (int r = 0; r < 8; ++r) {
      float s = 0.f;
#pragma unroll
      for (int w = 0; w < 8; ++w) s += sRedSum[w * 16 + 8 * g + r];
      rinv[r] = 1.0f / s;
    }
    if (wave == 0 && ln == 0) {
#pragma unroll
      for (int r = 0; r < 8; ++r) sRowInv[8 * g + r] = rinv[r];
    }

    // -------- store p (bf16) for att@V; accumulate head-mean numerator ------
#pragma unroll
    for (int kt = 0; kt < 8; ++kt) {
      const int key = (wave * 8 + kt) * 16 + ln;
#pragma unroll
      for (int r = 0; r < 8; ++r) {
        const int row = 8 * g + r;
        const float p = sc[kt][r];
        sP[row * 1024 + key] = (bf16)p;
        sAtt[row * 1024 + key] += p * rinv[r];
      }
    }
    __syncthreads();

    // -------- y_h = P @ V_h : wave = (ntile 0..3) x (khalf 0..1) --------
    {
      const int ntile = wave & 3;
      const int khalf = wave >> 2;
      v8f ya = zero8();
      const bf16* vrow =
          Vt + ((size_t)(bIdx * 1024 + h * 64 + ntile * 16 + ln)) * 1024;
      const bf16* prow = sP + ln * 1024;
      const int kbeg = khalf * 512;
      for (int kk = kbeg; kk < kbeg + 512; kk += 32) {
        __builtin_prefetch(vrow + kk + 128, 0, 0);
        v16bf pa = load_ab(prow + kk, g);
        v16bf vb = load_ab(vrow + kk, g);
        ya = wmma_bf16(pa, vb, ya);
      }
#pragma unroll
      for (int r = 0; r < 8; ++r)
        sYp[(khalf * 16 + 8 * g + r) * 64 + ntile * 16 + ln] = ya[r];
    }
    __syncthreads();

    // -------- combine K-halves, normalize rows, emit bf16 y for projection --
    for (int idx = tid; idx < 1024; idx += 256) {
      const int m = idx >> 6, c = idx & 63;
      const float v = (sYp[m * 64 + c] + sYp[(16 + m) * 64 + c]) * sRowInv[m];
      Ytmp[((size_t)(bIdx * 1024 + t0 + m)) * 1024 + h * 64 + c] = (bf16)v;
    }
    __syncthreads();
  }

  // -------- head-mean attention output --------
  {
    const int m  = tid >> 4;
    const int cb = (tid & 15) * 64;
    const float sc16 = 1.0f / 16.0f;
    float* dst = attOut + ((size_t)(bIdx * 1024 + t0 + m)) * 1024 + cb;
    for (int i = 0; i < 64; i += 4) {
      v4f v = *(v4f*)(sAtt + m * 1024 + cb + i);
      v4f o = {v[0] * sc16, v[1] * sc16, v[2] * sc16, v[3] * sc16};
      *(v4f*)(dst + i) = o;
    }
  }
}

// ---------------------------------------------------------------------------
extern "C" void kernel_launch(void* const* d_in, const int* in_sizes, int n_in,
                              void* d_out, int out_size, void* d_ws,
                              size_t ws_size, hipStream_t stream) {
  (void)in_sizes; (void)n_in; (void)out_size; (void)ws_size;

  const float* q_in = (const float*)d_in[0];
  const float* k_in = (const float*)d_in[1];
  const float* v_in = (const float*)d_in[2];
  const float* Wq   = (const float*)d_in[3];
  const float* bq   = (const float*)d_in[4];
  const float* Wk   = (const float*)d_in[5];
  const float* bk   = (const float*)d_in[6];
  const float* Wv   = (const float*)d_in[7];
  const float* bv   = (const float*)d_in[8];
  const float* Wo   = (const float*)d_in[9];
  const float* bo   = (const float*)d_in[10];

  float* yOut   = (float*)d_out;                       // [4,1024,1024]
  float* attOut = yOut + (size_t)4 * 1024 * 1024;      // [4,1024,1024]

  // Workspace carve (all bf16): 3 inputs, 4 weights, Q, K, Vt, Ytmp = 64 MB.
  char* ws = (char*)d_ws;
  const size_t XSZ = (size_t)4096 * 1024 * sizeof(bf16); // 8 MB
  const size_t WSZ = (size_t)1024 * 1024 * sizeof(bf16); // 2 MB
  bf16* Xq  = (bf16*)(ws);
  bf16* Xk  = (bf16*)(ws + XSZ);
  bf16* Xv  = (bf16*)(ws + 2 * XSZ);
  bf16* Wqb = (bf16*)(ws + 3 * XSZ);
  bf16* Wkb = (bf16*)(ws + 3 * XSZ + WSZ);
  bf16* Wvb = (bf16*)(ws + 3 * XSZ + 2 * WSZ);
  bf16* Wob = (bf16*)(ws + 3 * XSZ + 3 * WSZ);
  bf16* Qb  = (bf16*)(ws + 3 * XSZ + 4 * WSZ);
  bf16* Kb  = (bf16*)(ws + 4 * XSZ + 4 * WSZ);
  bf16* Vt  = (bf16*)(ws + 5 * XSZ + 4 * WSZ);
  bf16* Ytmp= (bf16*)(ws + 6 * XSZ + 4 * WSZ);

  // 1) bf16 conversions
  const int NIN = 4096 * 1024, NW = 1024 * 1024;
  cvt_bf16<<<NIN / 2048, 256, 0, stream>>>(q_in, Xq, NIN);
  cvt_bf16<<<NIN / 2048, 256, 0, stream>>>(k_in, Xk, NIN);
  cvt_bf16<<<NIN / 2048, 256, 0, stream>>>(v_in, Xv, NIN);
  cvt_bf16<<<NW / 2048, 256, 0, stream>>>(Wq, Wqb, NW);
  cvt_bf16<<<NW / 2048, 256, 0, stream>>>(Wk, Wkb, NW);
  cvt_bf16<<<NW / 2048, 256, 0, stream>>>(Wv, Wvb, NW);
  cvt_bf16<<<NW / 2048, 256, 0, stream>>>(Wo, Wob, NW);

  // 2) QKV projections (WMMA bf16 + TDM-staged W tiles): grid = (N/64, M/128)
  dim3 gg(16, 32);
  gemm_xwT<1><<<gg, 256, 0, stream>>>(Xq, Wqb, bq, (void*)Qb, 4096, 1024, 1024);
  gemm_xwT<1><<<gg, 256, 0, stream>>>(Xk, Wkb, bk, (void*)Kb, 4096, 1024, 1024);
  gemm_xwT<2><<<gg, 256, 0, stream>>>(Xv, Wvb, bv, (void*)Vt, 4096, 1024, 1024);

  // 3) attention core: 256 workgroups (4 batches x 64 query blocks)
  (void)hipFuncSetAttribute((const void*)attention_kernel,
                            hipFuncAttributeMaxDynamicSharedMemorySize,
                            SM_TOTAL);
  attention_kernel<<<256, 256, SM_TOTAL, stream>>>(Qb, Kb, Vt, Ytmp, attOut);

  // 4) output projection -> f32 y
  gemm_xwT<0><<<gg, 256, 0, stream>>>(Ytmp, Wob, bo, (void*)yOut, 4096, 1024, 1024);
}
